// SpatialStyleModLayer_46935402611210
// MI455X (gfx1250) — matrix operand order repaired and checked by reading, not compiled, not run on hardware
//
#include <hip/hip_runtime.h>

// ---------------------------------------------------------------------------
// SpatialStyleModLayer for MI455X (gfx1250, wave32, WMMA bf16 path)
//
// Roofline: 103 GFLOP vs 768 MB HBM streams => 134 FLOP/B; only the bf16 WMMA
// (16x16x32, f32 accum) path clears the ~3.1 PFLOPS needed to stay HBM-bound
// at 23.3 TB/s (~33us). This revision targets the secondary bottleneck, L2
// bandwidth for weight-matrix (B) fragments: 2 row-tiles per workgroup share
// every B fragment, and the squared-weight fragment is computed in registers
// from the wT fragment instead of being re-streamed. B traffic drops from
// ~6.3 GB (~190 TB/s demand) to ~2.1 GB (~63 TB/s). Streaming tensors use
// non-temporal hints so they never pressure the hot 256 KB of weights in L2.
// ---------------------------------------------------------------------------

typedef __bf16 bf16_t;
typedef __attribute__((ext_vector_type(16))) __bf16 v16bf;
typedef __attribute__((ext_vector_type(8)))  __bf16 v8bf;
typedef __attribute__((ext_vector_type(8)))  float  v8f;
typedef __attribute__((ext_vector_type(4)))  float  v4f;

#define CH    256      // in/out/style channels
#define LDSK  264      // padded LDS row stride in bf16 elems (528B = 33*16B:
                       // 16B-aligned b128 ds ops, rows skewed across banks)
#define ROW_T 2        // row tiles (of 16 rows) per workgroup

// --- prep: bf16 copies of the constant matrices (L2-resident, 256KB) -------
//   affB[i][s] = bf16(affine_w[i][s])   (already B-column-major for GEMM1)
//   wT  [o][i] = bf16(weight[i][o])     (B-column-major for GEMM2/3)
__global__ __launch_bounds__(256) void prep_kernel(
    const float* __restrict__ weight, const float* __restrict__ affine_w,
    bf16_t* __restrict__ affB, bf16_t* __restrict__ wT)
{
  int idx = blockIdx.x * 256 + threadIdx.x;      // 0..65535
  int r = idx >> 8, c = idx & 255;               // r = o, c = i for wT
  affB[idx] = (bf16_t)affine_w[idx];
  wT[idx]   = (bf16_t)weight[c * 256 + r];       // weight[i][o]
}

// A-fragment (16x32 bf16) from padded LDS tile, contiguous K order (matches B):
// lane<16: row=lane, K in kb*32+0..15; lane>=16: row=lane-16,
// K in kb*32+16..31. Two aligned 16B ds loads.
__device__ inline v16bf afrag_lds(const bf16_t* base, int lane, int kb) {
  int row = lane & 15;
  int k0  = kb * 32 + (lane >> 4) * 16;
  const bf16_t* p = base + row * LDSK + k0;
  v8bf a0 = *(const v8bf*)p;
  v8bf a1 = *(const v8bf*)(p + 8);
  return __builtin_shufflevector(a0, a1, 0,1,2,3,4,5,6,7,8,9,10,11,12,13,14,15);
}

// B-fragment (32x16 bf16) from a column-major-for-B global matrix mat[col][k]:
// lane holds col = colBase + (lane&15), K = kb*32 + 16*(lane>=16) + [0,16).
// One contiguous 32B load per lane (hot data: default temporal hints).
__device__ inline v16bf bfrag_glb(const bf16_t* __restrict__ mat,
                                  int colBase, int lane, int kb) {
  int col = colBase + (lane & 15);
  int k0  = kb * 32 + (lane >> 4) * 16;
  return *(const v16bf*)(mat + col * 256 + k0);
}

// Elementwise square of a bf16 B fragment (for ssq GEMM): hidden under the
// matrix pipe since XDL WMMAs co-execute with VALU.
__device__ inline v16bf sq16(v16bf b) {
  v16bf r;
#pragma unroll
  for (int j = 0; j < 16; ++j) { float f = (float)b[j]; r[j] = (bf16_t)(f * f); }
  return r;
}

// Main kernel: 512 threads = 16 waves; each workgroup owns 32 rows x 256 cols
// (2 row tiles). Wave w owns output columns [16w, 16w+16) for both tiles.
__global__ __launch_bounds__(512) void modlinear_kernel(
    const float* __restrict__ x, const float* __restrict__ style,
    const float* __restrict__ bias, const float* __restrict__ affine_b,
    const bf16_t* __restrict__ affB, const bf16_t* __restrict__ wT,
    float* __restrict__ out)
{
  __shared__ bf16_t sA [ROW_T][16 * LDSK];   // style tiles, bf16
  __shared__ bf16_t xmA[ROW_T][16 * LDSK];   // x*mod tiles, bf16
  __shared__ bf16_t m2A[ROW_T][16 * LDSK];   // mod^2 tiles, bf16

  const int  t    = threadIdx.x;
  const int  lane = t & 31;
  const int  wv   = t >> 5;                          // wave id 0..15
  const long rowB = (long)blockIdx.x * (16 * ROW_T); // first row of this block

  // ---- stage style tiles [2][16 x 256] f32 -> bf16 LDS (coalesced, NT) ----
  {
    int li = t * 8;                                  // 0..4095
    int r = li >> 8, c = li & 255;
#pragma unroll
    for (int tt = 0; tt < ROW_T; ++tt) {
      const float* gp = style + (rowB + tt * 16 + r) * 256 + c;
      v4f s0 = __builtin_nontemporal_load((const v4f*)gp);
      v4f s1 = __builtin_nontemporal_load((const v4f*)(gp + 4));
      v8bf bb;
#pragma unroll
      for (int j = 0; j < 4; ++j) { bb[j] = (bf16_t)s0[j]; bb[4 + j] = (bf16_t)s1[j]; }
      *(v8bf*)(&sA[tt][0] + r * LDSK + c) = bb;
    }
  }

  const int colBase = wv * 16;
  const int col     = colBase + (lane & 15);   // this lane's output column
  const int hi      = lane >> 4;               // C-layout row offset select

  // ---- prefetch x values for both tiles (latency overlaps GEMM1) ----------
  float xv[ROW_T][8];
#pragma unroll
  for (int tt = 0; tt < ROW_T; ++tt)
#pragma unroll
    for (int r = 0; r < 8; ++r) {
      int row = tt * 16 + r + hi * 8;
      xv[tt][r] = __builtin_nontemporal_load(x + (rowB + row) * 256 + col);
    }

  __syncthreads();

  // ---- GEMM1: mod = style @ affine_w^T (K=256); B shared across row tiles -
  v8f mAcc[ROW_T] = {};
#pragma unroll
  for (int kb = 0; kb < 8; ++kb) {
    v16bf b = bfrag_glb(affB, colBase, lane, kb);
#pragma unroll
    for (int tt = 0; tt < ROW_T; ++tt) {
      v16bf a = afrag_lds(&sA[tt][0], lane, kb);
      mAcc[tt] = __builtin_amdgcn_wmma_f32_16x16x32_bf16(false, a, false, b,
                                                         (short)0, mAcc[tt],
                                                         false, false);
    }
  }
  const float ab1 = affine_b[col] + 1.0f;

  // ---- elementwise: xm = x*mod, m2 = mod^2 -> bf16 LDS (C layout M=r+8*hi)
#pragma unroll
  for (int tt = 0; tt < ROW_T; ++tt)
#pragma unroll
    for (int r = 0; r < 8; ++r) {
      int   row = tt * 16 + r + hi * 8;
      float md  = mAcc[tt][r] + ab1;
      xmA[0][row * LDSK + col] = (bf16_t)(xv[tt][r] * md);  // [tt] folded in row
      m2A[0][row * LDSK + col] = (bf16_t)(md * md);
    }
  __syncthreads();

  // ---- GEMM2: out0 = xm @ w ; GEMM3: ssq = m2 @ w^2 (w^2 from registers) --
  v8f oAcc[ROW_T] = {};
  v8f qAcc[ROW_T] = {};
#pragma unroll
  for (int kb = 0; kb < 8; ++kb) {
    v16bf b  = bfrag_glb(wT, colBase, lane, kb);
    v16bf b2 = sq16(b);
#pragma unroll
    for (int tt = 0; tt < ROW_T; ++tt) {
      v16bf a1 = afrag_lds(&xmA[tt][0], lane, kb);
      oAcc[tt] = __builtin_amdgcn_wmma_f32_16x16x32_bf16(false, a1, false, b,
                                                         (short)0, oAcc[tt],
                                                         false, false);
      v16bf a2 = afrag_lds(&m2A[tt][0], lane, kb);
      qAcc[tt] = __builtin_amdgcn_wmma_f32_16x16x32_bf16(false, a2, false, b2,
                                                         (short)0, qAcc[tt],
                                                         false, false);
    }
  }

  // ---- demodulate + bias, non-temporal store ------------------------------
  const float bv = bias[col];
#pragma unroll
  for (int tt = 0; tt < ROW_T; ++tt)
#pragma unroll
    for (int r = 0; r < 8; ++r) {
      int row = tt * 16 + r + hi * 8;
      float v = oAcc[tt][r] * rsqrtf(qAcc[tt][r] + 1e-8f) + bv;
      __builtin_nontemporal_store(v, out + (rowB + row) * 256 + col);
    }
}

extern "C" void kernel_launch(void* const* d_in, const int* in_sizes, int n_in,
                              void* d_out, int out_size, void* d_ws, size_t ws_size,
                              hipStream_t stream) {
  // setup_inputs order: x, style, weight, bias, affine_w, affine_b (all f32)
  const float* x        = (const float*)d_in[0];
  const float* style    = (const float*)d_in[1];
  const float* weight   = (const float*)d_in[2];
  const float* bias     = (const float*)d_in[3];
  const float* affine_w = (const float*)d_in[4];
  const float* affine_b = (const float*)d_in[5];
  float*       out      = (float*)d_out;

  // workspace: 2 x 256x256 bf16 matrices = 256KB (L2-resident constants)
  bf16_t* affB = (bf16_t*)d_ws;
  bf16_t* wT   = affB + 256 * 256;

  prep_kernel<<<256, 256, 0, stream>>>(weight, affine_w, affB, wT);

  const long rows   = (long)out_size / 256;          // B*N = 262144
  const int  blocks = (int)(rows / (16 * ROW_T));    // 8192 blocks of 32 rows
  modlinear_kernel<<<blocks, 512, 0, stream>>>(x, style, bias, affine_b,
                                               affB, wT, out);
}
